// Block_68822555951557
// MI455X (gfx1250) — compile-verified, matrix operand order checked by reading, not run on hardware
//
#include <hip/hip_runtime.h>
#include <hip/hip_bf16.h>

// ---------------------------------------------------------------------------
// Differential-attention transformer block for MI455X (gfx1250, wave32).
// - All matmuls: v_wmma_f32_16x16x32_f16 (f16 operands, fp32 accumulate).
// - Shared tiles (GEMM B, attention K/V) staged into LDS with CDNA5
//   global_load_async_to_lds_b128 (ASYNCcnt) + double buffering, so each
//   workgroup fetches them once instead of once per wave.
// - Attention is flash-style (dual-stream online softmax); the 2048x2048
//   score matrices are never materialized.
// ---------------------------------------------------------------------------

typedef __attribute__((ext_vector_type(16))) _Float16 v16h;
typedef __attribute__((ext_vector_type(8)))  _Float16 v8h;
typedef __attribute__((ext_vector_type(8)))  float    v8f;

#define BATCH 2
#define NDQ   2048
#define NKV   2048
#define DIM   256
#define NH    8
#define HD    32
#define DQK   512
#define DFF   1024

// ---- WMMA helper -----------------------------------------------------------
__device__ __forceinline__ v8f wmma_f16(v16h a, v16h b, v8f c) {
  return __builtin_amdgcn_wmma_f32_16x16x32_f16(
      /*neg_a=*/false, a, /*neg_b=*/false, b,
      /*c_mod=*/(short)0, c, /*reuse_a=*/false, /*reuse_b=*/false);
}

// ---- CDNA5 async global->LDS copy (ISA 15.18.3, ASYNCcnt) ------------------
// Generic LDS addresses carry the LDS byte offset in addr[31:0] (ISA 10.2),
// so the low 32 bits of a __shared__ pointer are the VDST LDS offset.
__device__ __forceinline__ unsigned lds_off(const void* p) {
  return (unsigned)(size_t)p;
}
__device__ __forceinline__ void async_ld_b128(unsigned lds, const _Float16* g) {
  asm volatile("global_load_async_to_lds_b128 %0, %1, off"
               :: "v"(lds), "v"((unsigned long long)(size_t)g)
               : "memory");
}
#define WAIT_ASYNC(imm) asm volatile("s_wait_asynccnt " #imm ::: "memory")

// ---- 16x32 f16 A/B fragment loader (ISA 7.12.2 layout) --------------------
// lane L (L<16):  row r=L,    halves 0..7 = K 0..7,  halves 8..15 = K 16..23
// lane L (L>=16): row r=L-16, halves 0..7 = K 8..15, halves 8..15 = K 24..31
__device__ __forceinline__ v16h load_frag(const _Float16* __restrict__ base,
                                          int ldk, int lane) {
  int r  = lane & 15;
  int kb = (lane >> 4) << 3;           // 0 or 8
  const _Float16* p = base + (long)r * ldk + kb;
  v8h lo = *(const v8h*)(p);
  v8h hi = *(const v8h*)(p + 16);
  v16h out;
#pragma unroll
  for (int i = 0; i < 8; ++i) { out[i] = lo[i]; out[i + 8] = hi[i]; }
  return out;
}

// ---- half-wave (16-lane) row reductions for C-layout rows ------------------
__device__ __forceinline__ float hmax16(float x) {
#pragma unroll
  for (int o = 1; o < 16; o <<= 1) x = fmaxf(x, __shfl_xor(x, o, 32));
  return x;
}
__device__ __forceinline__ float hsum16(float x) {
#pragma unroll
  for (int o = 1; o < 16; o <<= 1) x += __shfl_xor(x, o, 32);
  return x;
}

// ---------------------------------------------------------------------------
// fp32 -> f16 convert
// ---------------------------------------------------------------------------
__global__ void cvt_f16(const float* __restrict__ s, _Float16* __restrict__ d, int n) {
  int i = blockIdx.x * 256 + threadIdx.x;
  if (i < n) d[i] = (_Float16)s[i];
}

// ---------------------------------------------------------------------------
// lambda[h] = exp(sum lq1*lk1) - exp(sum lq2*lk2) + 0.1 ; one wave per head
// ---------------------------------------------------------------------------
__global__ void lam_kernel(const float* __restrict__ lq1, const float* __restrict__ lk1,
                           const float* __restrict__ lq2, const float* __restrict__ lk2,
                           float* __restrict__ lam) {
  int h = threadIdx.x >> 5, d = threadIdx.x & 31;
  float p1 = lq1[h * HD + d] * lk1[h * HD + d];
  float p2 = lq2[h * HD + d] * lk2[h * HD + d];
#pragma unroll
  for (int o = 1; o < 32; o <<= 1) { p1 += __shfl_xor(p1, o, 32); p2 += __shfl_xor(p2, o, 32); }
  if (d == 0) lam[h] = __expf(p1) - __expf(p2) + 0.1f;
}

// ---------------------------------------------------------------------------
// LayerNorm row of 256 -> f16  (one 256-thread block per row)
// ---------------------------------------------------------------------------
__global__ void layernorm_to_f16(const float* __restrict__ x, const float* __restrict__ g,
                                 const float* __restrict__ b, _Float16* __restrict__ out) {
  long row = blockIdx.x;
  int  t   = threadIdx.x;
  float v  = x[row * DIM + t];
  float s = v, ss = v * v;
#pragma unroll
  for (int o = 1; o < 32; o <<= 1) { s += __shfl_xor(s, o, 32); ss += __shfl_xor(ss, o, 32); }
  __shared__ float S[8], SS[8];
  int wv = t >> 5, ln = t & 31;
  if (ln == 0) { S[wv] = s; SS[wv] = ss; }
  __syncthreads();
  if (t == 0) {
    float a = 0.f, c = 0.f;
#pragma unroll
    for (int i = 0; i < 8; ++i) { a += S[i]; c += SS[i]; }
    S[0] = a; SS[0] = c;
  }
  __syncthreads();
  float mu  = S[0] * (1.0f / DIM);
  float var = SS[0] * (1.0f / DIM) - mu * mu;
  float y = (v - mu) * rsqrtf(var + 1e-5f) * g[t] + b[t];
  out[row * DIM + t] = (_Float16)y;
}

// ---------------------------------------------------------------------------
// Generic WMMA GEMM:  C[z][m][n] = act( A[z][m][:] . Bt[z][n][:] + bias[n] ) + res
// Wave tile 16(M) x 64(N); 4 waves -> 64x64 per workgroup. K multiple of 32.
// B tile (64x32 f16, shared by all 4 waves) is staged into LDS with async
// copies, double-buffered against the WMMA compute.
// ---------------------------------------------------------------------------
__global__ __launch_bounds__(128)
void gemm16(const _Float16* __restrict__ A, const _Float16* __restrict__ Bt,
            const float* __restrict__ bias, const float* __restrict__ res,
            _Float16* __restrict__ Ch, float* __restrict__ Cf,
            int M, int N, int K,
            long aZ, long bZ, long cZ, long rZ, int gelu) {
  __shared__ __attribute__((aligned(16))) _Float16 Bsh[2][64 * 32];
  int tid  = threadIdx.x;
  int wave = tid >> 5, lane = tid & 31;
  int z  = blockIdx.z;
  int m0 = blockIdx.x * 64 + wave * 16;
  int n0 = blockIdx.y * 64;
  const _Float16* Ap = A + aZ * z + (long)m0 * K;
  const _Float16* Bp = Bt + bZ * z + (long)n0 * K;

  // 64 rows x 64 bytes = 256 x 16B segments, 2 per thread.
  int s0 = tid, s1 = tid + 128;
  const _Float16* g0 = Bp + (long)(s0 >> 2) * K + ((s0 & 3) << 3);
  const _Float16* g1 = Bp + (long)(s1 >> 2) * K + ((s1 & 3) << 3);

  v8f acc0 = {}, acc1 = {}, acc2 = {}, acc3 = {};
  int nsteps = K >> 5;

  {
    unsigned base = lds_off(&Bsh[0][0]);
    async_ld_b128(base + s0 * 16, g0);
    async_ld_b128(base + s1 * 16, g1);
  }
  for (int ks = 0; ks < nsteps; ++ks) {
    int bf = ks & 1;
    if (ks + 1 < nsteps) {
      unsigned base = lds_off(&Bsh[bf ^ 1][0]);
      int k = (ks + 1) << 5;
      async_ld_b128(base + s0 * 16, g0 + k);
      async_ld_b128(base + s1 * 16, g1 + k);
      WAIT_ASYNC(0x2);                 // previous stage (in-order) complete
    } else {
      WAIT_ASYNC(0x0);
    }
    __syncthreads();

    __builtin_prefetch((const void*)(Ap + ks * 32 + 256), 0, 3);
    v16h a  = load_frag(Ap + (long)ks * 32, K, lane);
    v16h b0 = load_frag(&Bsh[bf][0],       32, lane);
    v16h b1 = load_frag(&Bsh[bf][16 * 32], 32, lane);
    v16h b2 = load_frag(&Bsh[bf][32 * 32], 32, lane);
    v16h b3 = load_frag(&Bsh[bf][48 * 32], 32, lane);
    acc0 = wmma_f16(a, b0, acc0);
    acc1 = wmma_f16(a, b1, acc1);
    acc2 = wmma_f16(a, b2, acc2);
    acc3 = wmma_f16(a, b3, acc3);
    __syncthreads();                   // protect Bsh[bf] before restage
  }

  int nc = lane & 15, mb = (lane >> 4) << 3;
  long cz = cZ * z;
#pragma unroll
  for (int j = 0; j < 4; ++j) {
    v8f acc = (j == 0) ? acc0 : (j == 1) ? acc1 : (j == 2) ? acc2 : acc3;
#pragma unroll
    for (int v = 0; v < 8; ++v) {
      int m = m0 + v + mb;                 // C layout: row = v + 8*(lane>>4)
      int n = n0 + j * 16 + nc;            //           col = lane & 15
      float val = acc[v];
      if (bias) val += bias[n];
      if (gelu) val = 0.5f * val * (1.0f + erff(val * 0.70710678118f));
      if (res)  val += res[rZ * z + (long)m * N + n];
      long idx = cz + (long)m * N + n;
      if (Ch) Ch[idx] = (_Float16)val;
      if (Cf) Cf[idx] = val;
    }
  }
}

// ---------------------------------------------------------------------------
// Flash differential attention.
// Grid: (Nd/64, H, B), block 128 (4 waves). Each wave owns 16 query rows.
// K tile (32x64 f16) and V tile (32x32 f16), shared by all 4 waves, are
// async-staged into LDS, double-buffered. Per 32-key tile: 4 score WMMAs,
// dual-stream online softmax, P bounced via LDS (C->A layout), 4 P*V WMMAs.
// Final: O = O1/l1 - lam*O2/l2.
// ---------------------------------------------------------------------------
__global__ __launch_bounds__(128)
void diff_attn(const _Float16* __restrict__ Qh, const _Float16* __restrict__ Kh,
               const _Float16* __restrict__ Vt, const float* __restrict__ lam,
               _Float16* __restrict__ Oh) {
  const float SC = 0.17677669529663687f;   // HD^-0.5
  int tid  = threadIdx.x;
  int wave = tid >> 5, lane = tid & 31;
  int b = blockIdx.z, h = blockIdx.y;
  int q0 = blockIdx.x * 64 + wave * 16;
  int nc = lane & 15, mb = (lane >> 4) << 3;

  __shared__ __attribute__((aligned(16))) _Float16 Ksh[2][32 * 64];
  __shared__ __attribute__((aligned(16))) _Float16 Vsh[2][32 * 32];
  __shared__ __attribute__((aligned(16))) _Float16 pb[4][2][16 * 32];

  const _Float16* Qb = Qh + ((long)b * NDQ + q0) * DQK + h * 64;
  const _Float16* Kb = Kh + ((long)b * NKV) * DQK + h * 64;
  const _Float16* Vb = Vt + ((long)b * DIM + h * HD) * NKV;

  v16h q1 = load_frag(Qb, DQK, lane);
  v16h q2 = load_frag(Qb + HD, DQK, lane);

  v8f o1a = {}, o1b = {}, o2a = {}, o2b = {};
  float m1[8], l1[8], m2[8], l2[8];
#pragma unroll
  for (int v = 0; v < 8; ++v) { m1[v] = -1e30f; l1[v] = 0.f; m2[v] = -1e30f; l2[v] = 0.f; }

  // Stage one 32-key tile: K = 32 rows x 128B (256 segs, 2/thread),
  //                        V = 32 rows x  64B (128 segs, 1/thread).
  int s0 = tid, s1 = tid + 128;
  auto stage = [&](int kb, int bf) {
    unsigned kbase = lds_off(&Ksh[bf][0]);
    unsigned vbase = lds_off(&Vsh[bf][0]);
    async_ld_b128(kbase + s0 * 16, Kb + (long)(kb + (s0 >> 3)) * DQK + ((s0 & 7) << 3));
    async_ld_b128(kbase + s1 * 16, Kb + (long)(kb + (s1 >> 3)) * DQK + ((s1 & 7) << 3));
    async_ld_b128(vbase + tid * 16, Vb + (long)(tid >> 2) * NKV + kb + ((tid & 3) << 3));
  };

  stage(0, 0);
  for (int it = 0; it < NKV / 32; ++it) {
    int kb = it * 32;
    int bf = it & 1;
    if (kb + 32 < NKV) { stage(kb + 32, bf ^ 1); WAIT_ASYNC(0x3); }
    else               { WAIT_ASYNC(0x0); }
    __syncthreads();

    v16h k1a = load_frag(&Ksh[bf][0],            64, lane);
    v16h k1b = load_frag(&Ksh[bf][16 * 64],      64, lane);
    v16h k2a = load_frag(&Ksh[bf][32],           64, lane);
    v16h k2b = load_frag(&Ksh[bf][16 * 64 + 32], 64, lane);

    v8f z = {};
    v8f s1l = wmma_f16(q1, k1a, z);
    v8f s1h = wmma_f16(q1, k1b, z);
    v8f s2l = wmma_f16(q2, k2a, z);
    v8f s2h = wmma_f16(q2, k2b, z);

#pragma unroll
    for (int v = 0; v < 8; ++v) {
      int rr = v + mb;                      // row within the 16-row q tile
      // ---- stream 1 ----
      float a0 = s1l[v] * SC, a1 = s1h[v] * SC;
      float mt = hmax16(fmaxf(a0, a1));
      float mn = fmaxf(m1[v], mt);
      float cr = __expf(m1[v] - mn);
      float p0 = __expf(a0 - mn), p1 = __expf(a1 - mn);
      l1[v] = l1[v] * cr + hsum16(p0 + p1);
      o1a[v] *= cr; o1b[v] *= cr;
      m1[v] = mn;
      pb[wave][0][rr * 32 + nc]      = (_Float16)p0;
      pb[wave][0][rr * 32 + nc + 16] = (_Float16)p1;
      // ---- stream 2 ----
      float c0 = s2l[v] * SC, c1 = s2h[v] * SC;
      float mt2 = hmax16(fmaxf(c0, c1));
      float mn2 = fmaxf(m2[v], mt2);
      float cr2 = __expf(m2[v] - mn2);
      float u0 = __expf(c0 - mn2), u1 = __expf(c1 - mn2);
      l2[v] = l2[v] * cr2 + hsum16(u0 + u1);
      o2a[v] *= cr2; o2b[v] *= cr2;
      m2[v] = mn2;
      pb[wave][1][rr * 32 + nc]      = (_Float16)u0;
      pb[wave][1][rr * 32 + nc + 16] = (_Float16)u1;
    }
    __syncthreads();

    v16h pa1 = load_frag(&pb[wave][0][0], 32, lane);
    v16h pa2 = load_frag(&pb[wave][1][0], 32, lane);
    v16h vb0 = load_frag(&Vsh[bf][0],       32, lane);   // d = 0..15
    v16h vb1 = load_frag(&Vsh[bf][16 * 32], 32, lane);   // d = 16..31
    o1a = wmma_f16(pa1, vb0, o1a);
    o1b = wmma_f16(pa1, vb1, o1b);
    o2a = wmma_f16(pa2, vb0, o2a);
    o2b = wmma_f16(pa2, vb1, o2b);
    __syncthreads();                    // protect Ksh/Vsh[bf] before restage
  }

  float la = lam[h];
#pragma unroll
  for (int v = 0; v < 8; ++v) {
    int m = q0 + v + mb;
    long row = ((long)b * NDQ + m) * DIM + h * HD;
    float x0 = o1a[v] / l1[v] - la * (o2a[v] / l2[v]);
    float x1 = o1b[v] / l1[v] - la * (o2b[v] / l2[v]);
    Oh[row + nc]      = (_Float16)x0;
    Oh[row + 16 + nc] = (_Float16)x1;
  }
}

// ---------------------------------------------------------------------------
// Host launcher
// ---------------------------------------------------------------------------
extern "C" void kernel_launch(void* const* d_in, const int* in_sizes, int n_in,
                              void* d_out, int out_size, void* d_ws, size_t ws_size,
                              hipStream_t stream) {
  const float* drift = (const float*)d_in[0];
  const float* ocean = (const float*)d_in[1];
  const float* Wq    = (const float*)d_in[2];
  const float* Wk    = (const float*)d_in[3];
  const float* Wv    = (const float*)d_in[4];
  const float* Wo    = (const float*)d_in[5];
  const float* lq1   = (const float*)d_in[6];
  const float* lk1   = (const float*)d_in[7];
  const float* lq2   = (const float*)d_in[8];
  const float* lk2   = (const float*)d_in[9];
  const float* gamma = (const float*)d_in[10];
  const float* beta  = (const float*)d_in[11];
  const float* fc1w  = (const float*)d_in[12];
  const float* fc1b  = (const float*)d_in[13];
  const float* fc2w  = (const float*)d_in[14];
  const float* fc2b  = (const float*)d_in[15];

  const long M  = (long)BATCH * NDQ;       // 4096 query rows
  const long Mo = (long)BATCH * NKV;       // 4096 key rows

  // ---- carve workspace -----------------------------------------------------
  size_t off = 0;
  auto carve = [&](size_t bytes) -> char* {
    char* p = (char*)d_ws + off;
    off += (bytes + 255) & ~(size_t)255;
    return p;
  };
  _Float16* Wqh  = (_Float16*)carve((size_t)DQK * DIM * 2);
  _Float16* Wkh  = (_Float16*)carve((size_t)DQK * DIM * 2);
  _Float16* Wvh  = (_Float16*)carve((size_t)DIM * DIM * 2);
  _Float16* Woh  = (_Float16*)carve((size_t)DIM * DIM * 2);
  _Float16* f1h  = (_Float16*)carve((size_t)DFF * DIM * 2);
  _Float16* f2h  = (_Float16*)carve((size_t)DIM * DFF * 2);
  float*    lamw = (float*)   carve(NH * 4);
  _Float16* xh   = (_Float16*)carve((size_t)M * DIM * 2);
  _Float16* oh   = (_Float16*)carve((size_t)Mo * DIM * 2);
  _Float16* Qhb  = (_Float16*)carve((size_t)M * DQK * 2);
  _Float16* Khb  = (_Float16*)carve((size_t)Mo * DQK * 2);
  _Float16* Vth  = (_Float16*)carve((size_t)BATCH * DIM * NKV * 2);
  _Float16* Ah   = (_Float16*)carve((size_t)M * DIM * 2);
  float*    dr2  = (float*)   carve((size_t)M * DIM * 4);
  _Float16* yh   = (_Float16*)carve((size_t)M * DIM * 2);
  _Float16* hh   = (_Float16*)carve((size_t)M * DFF * 2);
  (void)ws_size; (void)n_in; (void)in_sizes; (void)out_size;

  // ---- weight / input conversion ------------------------------------------
  auto cvt = [&](const float* s, _Float16* d, int n) {
    cvt_f16<<<(n + 255) / 256, 256, 0, stream>>>(s, d, n);
  };
  cvt(Wq,   Wqh, DQK * DIM);
  cvt(Wk,   Wkh, DQK * DIM);
  cvt(Wv,   Wvh, DIM * DIM);
  cvt(Wo,   Woh, DIM * DIM);
  cvt(fc1w, f1h, DFF * DIM);
  cvt(fc2w, f2h, DIM * DFF);
  cvt(ocean, oh, (int)(Mo * DIM));
  lam_kernel<<<1, 256, 0, stream>>>(lq1, lk1, lq2, lk2, lamw);

  // ---- LN1: x = LN(drift) --------------------------------------------------
  layernorm_to_f16<<<(int)M, 256, 0, stream>>>(drift, gamma, beta, xh);

  // ---- projections ---------------------------------------------------------
  // Q = x @ Wq^T   [4096 x 512]
  gemm16<<<dim3(M / 64, DQK / 64, 1), 128, 0, stream>>>(
      xh, Wqh, nullptr, nullptr, Qhb, nullptr, (int)M, DQK, DIM, 0, 0, 0, 0, 0);
  // K = ocean @ Wk^T   [4096 x 512]
  gemm16<<<dim3(Mo / 64, DQK / 64, 1), 128, 0, stream>>>(
      oh, Wkh, nullptr, nullptr, Khb, nullptr, (int)Mo, DQK, DIM, 0, 0, 0, 0, 0);
  // V^T[b] = Wv @ ocean[b]^T   [256 x 2048] per batch (z = batch)
  gemm16<<<dim3(DIM / 64, NKV / 64, BATCH), 128, 0, stream>>>(
      Wvh, oh, nullptr, nullptr, Vth, nullptr, DIM, NKV, DIM,
      0, (long)NKV * DIM, (long)DIM * NKV, 0, 0);

  // ---- flash differential attention ---------------------------------------
  diff_attn<<<dim3(NDQ / 64, NH, BATCH), 128, 0, stream>>>(Qhb, Khb, Vth, lamw, Ah);

  // ---- output projection + residual: drift2 = drift + attn @ Wo^T ----------
  gemm16<<<dim3(M / 64, DIM / 64, 1), 128, 0, stream>>>(
      Ah, Woh, nullptr, drift, nullptr, dr2, (int)M, DIM, DIM, 0, 0, 0, 0, 0);

  // ---- LN2 + FFN -----------------------------------------------------------
  layernorm_to_f16<<<(int)M, 256, 0, stream>>>(dr2, gamma, beta, yh);
  // h = gelu(y @ fc1^T + b1)   [4096 x 1024]
  gemm16<<<dim3(M / 64, DFF / 64, 1), 128, 0, stream>>>(
      yh, f1h, fc1b, nullptr, hh, nullptr, (int)M, DFF, DIM, 0, 0, 0, 0, 1);
  // out = drift2 + h @ fc2^T + b2  -> d_out (fp32, first half)
  gemm16<<<dim3(M / 64, DIM / 64, 1), 128, 0, stream>>>(
      hh, f2h, fc2b, dr2, nullptr, (float*)d_out, (int)M, DIM, DFF, 0, 0, 0, 0, 0);

  // ---- ocean passthrough ---------------------------------------------------
  hipMemcpyAsync((float*)d_out + (size_t)M * DIM, ocean,
                 (size_t)Mo * DIM * sizeof(float),
                 hipMemcpyDeviceToDevice, stream);
}
